// Solver_85487029059719
// MI455X (gfx1250) — matrix operand order repaired and checked by reading, not compiled, not run on hardware
//
#include <hip/hip_runtime.h>
#include <hip/hip_bf16.h>

typedef __attribute__((ext_vector_type(16))) __bf16 v16bf;
typedef __attribute__((ext_vector_type(8)))  float  v8f;

#define ZD   128
#define HIDD 245
#define ND   466
#define PD   4
#define TEV  128
#define NWG  16          // 16 workgroups, one per WGP, each owns 8 j-values of B
#define JPW  8
#define NKT  33          // 32 B k-tiles + 1 A-extension tile per i-tile
#define NIT  8           // 128 / 16 i-tiles
#define FRAG_HALVES 512  // 32 lanes * 16 bf16 halves = 1 KB fragment
#define SLICE_HALVES (NIT * NKT * FRAG_HALVES)   // 135168 halves
#define SLICE_BYTES  (SLICE_HALVES * 2)          // 270336 bytes
#define ODE_LDS (SLICE_BYTES + 4 * ZD * 9)       // + zv, argf, ypart, kbuf[6]

// workspace layout (bytes)
#define WS_BWS   0
#define WS_Z0    (NWG * (size_t)SLICE_BYTES)          // 4,325,376
#define WS_ZS    (WS_Z0 + 512)
#define WS_GPART (WS_ZS + (size_t)TEV * ZD * 4)
#define WS_CNT   (WS_GPART + (size_t)2 * NWG * ZD * 4)

__constant__ float ACO[5][5] = {
  {0.2f, 0.f, 0.f, 0.f, 0.f},
  {3.f/40.f, 9.f/40.f, 0.f, 0.f, 0.f},
  {44.f/45.f, -56.f/15.f, 32.f/9.f, 0.f, 0.f},
  {19372.f/6561.f, -25360.f/2187.f, 64448.f/6561.f, -212.f/729.f, 0.f},
  {9017.f/3168.f, -355.f/33.f, 46732.f/5247.f, 49.f/176.f, -5103.f/18656.f}};

// ---------------------------------------------------------------------------
// Prep: swizzle B (f32 [128][128][128]) + A (f32 [128][128]) into bf16 WMMA
// A-operand fragments. Fragment (it,kt) of WG w: lane L holds 16 halves at
// K(L,h) = h + (L<16 ? 0 : 16), row m = L%16 (contiguous k order, 16-bit
// A-matrix layout). kt<32 -> B[i][j0+kt/4][(kt%4)*32 + K]; kt==32 -> A-ext.
// ---------------------------------------------------------------------------
__global__ __launch_bounds__(256) void prep_kernel(const float* __restrict__ A,
                                                   const float* __restrict__ B,
                                                   __bf16* __restrict__ bws) {
  int idx = blockIdx.x * 256 + threadIdx.x;       // exact multiple, no guard
  int w  = idx / SLICE_HALVES;
  int r  = idx % SLICE_HALVES;
  int F  = r >> 9;
  int q  = r & 511;
  int L  = q >> 4;
  int h  = q & 15;
  int it = F / NKT;
  int kt = F % NKT;
  int m  = L & 15;
  int K  = h + (L & 16);
  int i  = it * 16 + m;
  float val;
  if (kt < 32) {
    int j = w * JPW + (kt >> 2);
    int k = ((kt & 3) << 5) + K;
    val = B[i * (ZD * ZD) + j * ZD + k];
  } else {
    val = (K < 8) ? A[i * ZD + w * JPW + K] : 0.0f;
  }
  bws[idx] = (__bf16)val;
}

// ---------------------------------------------------------------------------
// Encoder (batch=1) + barrier counter init + zs[0] = z0
// ---------------------------------------------------------------------------
__global__ __launch_bounds__(256) void enc_kernel(const float* __restrict__ n0,
                                                  const float* __restrict__ p,
                                                  const float* __restrict__ W1,
                                                  const float* __restrict__ b1,
                                                  const float* __restrict__ W2,
                                                  const float* __restrict__ b2,
                                                  float* __restrict__ z0out,
                                                  float* __restrict__ zs,
                                                  unsigned int* __restrict__ cnt) {
  __shared__ float x0[PD + ND];
  __shared__ float h1[HIDD];
  int tid = threadIdx.x;
  for (int i = tid; i < PD + ND; i += 256) x0[i] = (i < PD) ? p[i] : n0[i - PD];
  if (tid == 0) cnt[0] = 0u;
  __syncthreads();
  if (tid < HIDD) {
    float s = b1[tid];
    const float* wr = W1 + tid * (PD + ND);
    for (int i = 0; i < PD + ND; ++i) s += wr[i] * x0[i];
    h1[tid] = (s > 0.f) ? s : 0.2f * s;
  }
  __syncthreads();
  if (tid < ZD) {
    float s = b2[tid];
    const float* wr = W2 + tid * HIDD;
    for (int i = 0; i < HIDD; ++i) s += wr[i] * h1[i];
    float z = tanhf(s);
    z0out[tid] = z;
    zs[tid]   = z;
  }
}

// ---------------------------------------------------------------------------
// Persistent ODE kernel: 16 WGs, B-slice in LDS, WMMA bf16 matvec per stage,
// cross-WG reduction through L2 + generation barrier.
// ---------------------------------------------------------------------------
__global__ __launch_bounds__(256, 1) void ode_kernel(const float* __restrict__ tstep,
                                                     const __bf16* __restrict__ bws,
                                                     const float* __restrict__ z0,
                                                     float* __restrict__ zs,
                                                     float* gpart,
                                                     unsigned int* cnt) {
  extern __shared__ __align__(16) char smem[];
  __bf16* sl   = (__bf16*)smem;
  float*  zv   = (float*)(smem + SLICE_BYTES);
  float*  argf = zv + ZD;
  float*  ypart= argf + ZD;
  float*  kbuf = ypart + ZD;                     // 6 * ZD

  const int tid  = threadIdx.x;
  const int wg   = blockIdx.x;
  const int lane = tid & 31;
  const int wave = tid >> 5;
  const int j0   = wg * JPW;

  { // stage B-slice into LDS (one-time, 264 KB)
    const uint4* src = (const uint4*)(bws + (size_t)wg * SLICE_HALVES);
    uint4* dst = (uint4*)sl;
    for (int i = tid; i < SLICE_BYTES / 16; i += 256) dst[i] = src[i];
  }
  if (tid < ZD) zv[tid] = z0[tid];
  __syncthreads();

  unsigned int gen = 0;
  int parity = 0;

  for (int t = 1; t < TEV; ++t) {
    float dt = tstep[t] - tstep[t - 1];
    float hs = dt * 0.125f;
    for (int sub = 0; sub < 8; ++sub) {
      if (tid < ZD) { argf[tid] = zv[tid]; ypart[tid] = 0.f; }
      __syncthreads();
      for (int s = 0; s < 6; ++s) {
        // ---- build B-operands once per stage (N=0 column: lanes 0 & 16) ----
        v16bf bfr[4];
        for (int u = 0; u < 4; ++u) {
          int kt = wave + u * 8;
          int kb = ((kt & 3) << 5) + (lane & 16);
          float zj = argf[j0 + (kt >> 2)];
          v16bf b;
          if (lane == 0 || lane == 16) {
            for (int hh = 0; hh < 16; ++hh) b[hh] = (__bf16)(zj * argf[kb + hh]);
          } else {
            for (int hh = 0; hh < 16; ++hh) b[hh] = (__bf16)0.0f;
          }
          bfr[u] = b;
        }
        v16bf bext;
        {
          v16bf b;
          for (int hh = 0; hh < 16; ++hh) b[hh] = (__bf16)0.0f;
          if (lane == 0) for (int hh = 0; hh < 8; ++hh) b[hh] = (__bf16)argf[j0 + hh];
          bext = b;
        }
        // ---- WMMA over 8 i-tiles; wave v owns kt = v, v+8, v+16, v+24 ----
        for (int it = 0; it < NIT; ++it) {
          v8f acc = {};
          const int fb = it * NKT;
          for (int u = 0; u < 4; ++u) {
            int kt = wave + u * 8;
            const v16bf* ap = (const v16bf*)(sl + (fb + kt) * FRAG_HALVES + lane * 16);
            acc = __builtin_amdgcn_wmma_f32_16x16x32_bf16(
                false, ap[0], false, bfr[u], (short)0, acc, false, false);
          }
          if (it == wave) {  // A-extension tile (kt == 32), once per i-tile
            const v16bf* ap = (const v16bf*)(sl + (fb + 32) * FRAG_HALVES + lane * 16);
            acc = __builtin_amdgcn_wmma_f32_16x16x32_bf16(
                false, ap[0], false, bext, (short)0, acc, false, false);
          }
          // D layout: lane0 -> (M=r, N=0), lane16 -> (M=8+r, N=0)
          if (lane == 0) {
            for (int r = 0; r < 8; ++r) atomicAdd(&ypart[it * 16 + r], acc[r]);
          } else if (lane == 16) {
            for (int r = 0; r < 8; ++r) atomicAdd(&ypart[it * 16 + 8 + r], acc[r]);
          }
        }
        __syncthreads();
        // ---- publish partial (agent-visible) ----
        if (tid < ZD)
          __hip_atomic_store(gpart + parity * NWG * ZD + wg * ZD + tid, ypart[tid],
                             __ATOMIC_RELAXED, __HIP_MEMORY_SCOPE_AGENT);
        __syncthreads();
        // ---- generation barrier across 16 WGs ----
        if (tid == 0) {
          gen += NWG;
          __hip_atomic_fetch_add(cnt, 1u, __ATOMIC_RELEASE, __HIP_MEMORY_SCOPE_AGENT);
          while (__hip_atomic_load(cnt, __ATOMIC_ACQUIRE, __HIP_MEMORY_SCOPE_AGENT) < gen)
            __builtin_amdgcn_s_sleep(1);
        }
        __syncthreads();
        // ---- reduce partials, RK bookkeeping (redundant per WG) ----
        if (tid < ZD) {
          float kv = 0.f;
          float* gp = gpart + parity * NWG * ZD;
          for (int w2 = 0; w2 < NWG; ++w2)
            kv += __hip_atomic_load(gp + w2 * ZD + tid, __ATOMIC_RELAXED,
                                    __HIP_MEMORY_SCOPE_AGENT);
          kbuf[s * ZD + tid] = kv;
          if (s < 5) {
            float na = zv[tid];
            for (int r = 0; r <= s; ++r) na += hs * ACO[s][r] * kbuf[r * ZD + tid];
            argf[tid] = na;
          } else {
            zv[tid] = zv[tid] + hs * (35.f/384.f      * kbuf[0 * ZD + tid]
                                    + 500.f/1113.f    * kbuf[2 * ZD + tid]
                                    + 125.f/192.f     * kbuf[3 * ZD + tid]
                                    - 2187.f/6784.f   * kbuf[4 * ZD + tid]
                                    + 11.f/84.f       * kbuf[5 * ZD + tid]);
          }
          ypart[tid] = 0.f;
        }
        parity ^= 1;
        __syncthreads();
      }
    }
    if (wg == 0 && tid < ZD) zs[t * ZD + tid] = zv[tid];
  }
}

// ---------------------------------------------------------------------------
// Decoder: one WG per timestep
// ---------------------------------------------------------------------------
__global__ __launch_bounds__(256) void dec_kernel(const float* __restrict__ zs,
                                                  const float* __restrict__ W1,
                                                  const float* __restrict__ b1,
                                                  const float* __restrict__ W2,
                                                  const float* __restrict__ b2,
                                                  float* __restrict__ out) {
  __shared__ float zt[ZD];
  __shared__ float hd[HIDD];
  int tid = threadIdx.x, t = blockIdx.x;
  if (tid < ZD) zt[tid] = zs[t * ZD + tid];
  __syncthreads();
  if (tid < HIDD) {
    float s = b1[tid];
    const float* wr = W1 + tid * ZD;
    for (int i = 0; i < ZD; ++i) s += wr[i] * zt[i];
    hd[tid] = (s > 0.f) ? s : 0.2f * s;
  }
  __syncthreads();
  for (int o = tid; o < ND; o += 256) {
    float s = b2[o];
    const float* wr = W2 + o * HIDD;
    for (int i = 0; i < HIDD; ++i) s += wr[i] * hd[i];
    out[t * ND + o] = s;
  }
}

extern "C" void kernel_launch(void* const* d_in, const int* in_sizes, int n_in,
                              void* d_out, int out_size, void* d_ws, size_t ws_size,
                              hipStream_t stream) {
  (void)in_sizes; (void)n_in; (void)out_size; (void)ws_size;
  const float* n0  = (const float*)d_in[0];
  const float* p   = (const float*)d_in[1];
  const float* ts  = (const float*)d_in[2];
  const float* A   = (const float*)d_in[3];
  const float* B   = (const float*)d_in[4];
  const float* eW1 = (const float*)d_in[5];
  const float* eb1 = (const float*)d_in[6];
  const float* eW2 = (const float*)d_in[7];
  const float* eb2 = (const float*)d_in[8];
  const float* dW1 = (const float*)d_in[9];
  const float* db1 = (const float*)d_in[10];
  const float* dW2 = (const float*)d_in[11];
  const float* db2 = (const float*)d_in[12];

  char* ws = (char*)d_ws;
  __bf16*       bws = (__bf16*)(ws + WS_BWS);
  float*        z0  = (float*)(ws + WS_Z0);
  float*        zs  = (float*)(ws + WS_ZS);
  float*        gp  = (float*)(ws + WS_GPART);
  unsigned int* cnt = (unsigned int*)(ws + WS_CNT);

  (void)hipFuncSetAttribute((const void*)ode_kernel,
                            hipFuncAttributeMaxDynamicSharedMemorySize, ODE_LDS);

  prep_kernel<<<(NWG * SLICE_HALVES) / 256, 256, 0, stream>>>(A, B, bws);
  enc_kernel<<<1, 256, 0, stream>>>(n0, p, eW1, eb1, eW2, eb2, z0, zs, cnt);
  ode_kernel<<<NWG, 256, ODE_LDS, stream>>>(ts, bws, z0, zs, gp, cnt);
  dec_kernel<<<TEV, 256, 0, stream>>>(zs, dW1, db1, dW2, db2, (float*)d_out);
}
